// NERModel_21620865368112
// MI455X (gfx1250) — matrix operand order repaired and checked by reading, not compiled, not run on hardware
//
#include <hip/hip_runtime.h>
#include <hip/hip_bf16.h>

// Problem constants (match reference)
#define B_   256
#define T_   256
#define C_   20
#define V_   128
#define E_   50
#define H_   256
#define K_   25
#define G4H  1024
#define BT_  (B_ * T_)

typedef __attribute__((ext_vector_type(16))) __bf16        v16bf;
typedef __attribute__((ext_vector_type(8)))  float         v8f;
typedef __attribute__((ext_vector_type(4)))  unsigned int  u32x4;

union Frag {           // 32 bytes: one WMMA bf16 operand fragment per lane
    u32x4 q[2];
    v16bf v;
};

__device__ __forceinline__ unsigned short f2bf(float f) {
    union { float f; unsigned int u; } x; x.f = f;
    unsigned int r = x.u + 0x7fffu + ((x.u >> 16) & 1u);   // round-to-nearest-even
    return (unsigned short)(r >> 16);
}

// CDNA5 hardware transcendental: v_tanh_f32 (TRANS32 op).  The trailing
// s_delay_alu satisfies the ISA rule that a TRANS op needs one independent
// op before its output is consumed (compiler can't see inside the asm).
__device__ __forceinline__ float fast_tanh(float x) {
    float y;
    asm("v_tanh_f32 %0, %1\n\t"
        "s_delay_alu instid0(TRANS32_DEP_1)"
        : "=v"(y) : "v"(x));
    return y;
}
// sigmoid(x) = 0.5 + 0.5 * tanh(x/2): one trans op + two VALU ops.
__device__ __forceinline__ float fast_sigm(float x) {
    return fmaf(0.5f, fast_tanh(0.5f * x), 0.5f);
}

// ---------------------------------------------------------------------------
// Pre-swizzle weights (f32) into lane-major bf16 B-fragments.
// B-fragment layout chosen here (contiguous k, same as the A gather): lane L: n = nb*16 + (L&15),
// element j (0..15): k = kb*32 + (L>=16 ? 16 : 0) + j.  One frag = 1024 B,
// each lane owns 32 contiguous bytes -> two b128 loads per fragment.
// ---------------------------------------------------------------------------
__global__ void prep_frags(const float* __restrict__ W, unsigned short* __restrict__ dst,
                           int KB, int NB, int Kvalid, int Nvalid, int ldW, int transposed) {
    int total = KB * NB * 512;
    for (int idx = blockIdx.x * blockDim.x + threadIdx.x; idx < total;
         idx += gridDim.x * blockDim.x) {
        int frag   = idx >> 9;
        int within = idx & 511;
        int lane   = within >> 4;
        int j      = within & 15;
        int kb = frag / NB;
        int nb = frag - kb * NB;
        int n  = nb * 16 + (lane & 15);
        int k  = kb * 32 + ((lane >> 4) ? 16 : 0) + j;
        float v = 0.0f;
        if (k < Kvalid && n < Nvalid)
            v = transposed ? W[(size_t)n * ldW + k] : W[(size_t)k * ldW + n];
        dst[idx] = f2bf(v);
    }
}

__global__ void prep_embed(const float* __restrict__ emb, unsigned short* __restrict__ dst) {
    int idx = blockIdx.x * blockDim.x + threadIdx.x;
    if (idx < V_ * 64) {
        int v = idx >> 6, e = idx & 63;
        dst[idx] = f2bf(e < E_ ? emb[v * E_ + e] : 0.0f);
    }
}

// B fragment: coalesced 2x b128 per lane from pre-swizzled global stream.
__device__ __forceinline__ Frag load_bfrag(const u32x4* base, int kb, int NB, int nbg, int lane) {
    Frag f;
    const u32x4* p = base + (((size_t)(kb * NB + nbg)) << 6) + (lane << 1);
    f.q[0] = p[0];
    f.q[1] = p[1];
    return f;
}

// A fragment gather from LDS row-major bf16 tile (16 x rowStride).
// A k order used here (contiguous, matches prep_frags B): lane<16 holds K {0..15}, lane>=16 {16..31}.
__device__ __forceinline__ Frag load_afrag(const unsigned short* buf, int rowStride,
                                           int kb, int lane) {
    Frag f;
    int m  = lane & 15;
    int hi = lane >> 4;
    int k0 = kb * 32 + (hi ? 16 : 0);
    int k1 = kb * 32 + (hi ? 24 : 8);
    f.q[0] = *(const u32x4*)(buf + m * rowStride + k0);
    f.q[1] = *(const u32x4*)(buf + m * rowStride + k1);
    return f;
}

// ---------------------------------------------------------------------------
// Encoder: char LSTM (20 steps) + token LSTM step + classifier, all via
// v_wmma_f32_16x16x32_bf16.  One WG = 16 token rows, 8 waves; wave w owns
// hidden columns [w*32, w*32+32) across all four gates, so the LSTM pointwise
// update happens entirely in registers.
//
// K-outer accumulation: one live A-fragment, B-fragments streamed from the
// (L2-resident) pre-swizzled weight arrays with short live ranges -> fits in
// 256 VGPRs, no scratch spills.
// ---------------------------------------------------------------------------
__global__ __launch_bounds__(256)
void ner_encoder(const int* __restrict__ char_ids,      // [B,T,C]
                 const int* __restrict__ char_lengths,  // [B,T]
                 const float* __restrict__ b_c,         // [4H]
                 const float* __restrict__ b_t,         // [4H]
                 const float* __restrict__ b_cls,       // [K]
                 const unsigned short* __restrict__ wihc, // frags KB=2 NB=64
                 const unsigned short* __restrict__ whhc, // frags KB=8 NB=64
                 const unsigned short* __restrict__ wiht, // frags KB=8 NB=64
                 const unsigned short* __restrict__ wcls, // frags KB=8 NB=2
                 const unsigned short* __restrict__ embed,// [V][64] bf16
                 float* __restrict__ emis)                // [T][B][K]
{
    __shared__ unsigned short hbuf[16 * 256];  // h tile, bf16
    __shared__ unsigned short xbuf[16 * 64];   // x tile (E padded to 64), bf16
    __shared__ int lenbuf[16];

    const int tid     = threadIdx.x;
    const int lane    = tid & 31;
    const int wave    = tid >> 5;
    const int rowBase = blockIdx.x << 4;

    const u32x4* wihc4 = (const u32x4*)wihc;
    const u32x4* whhc4 = (const u32x4*)whhc;
    const u32x4* wiht4 = (const u32x4*)wiht;
    const u32x4* wcls4 = (const u32x4*)wcls;

    __builtin_prefetch(whhc4 + (wave << 7), 0, 1);   // global_prefetch_b8 (L2 warm)

    for (int i = tid; i < 16 * 256; i += 256) hbuf[i] = 0;
    if (tid < 16) lenbuf[tid] = char_lengths[rowBase + tid];

    // Per-thread pointwise ownership: (hf, r) -> row=(lane<16?r:r+8),
    // col = wave*32 + hf*16 + (lane&15).
    float bi[2], bff[2], bgg[2], boo[2];
#pragma unroll
    for (int hf = 0; hf < 2; ++hf) {
        int col = wave * 32 + hf * 16 + (lane & 15);
        bi[hf]  = b_c[col];
        bff[hf] = b_c[256 + col];
        bgg[hf] = b_c[512 + col];
        boo[hf] = b_c[768 + col];
    }
    float cc[2][8];
#pragma unroll
    for (int hf = 0; hf < 2; ++hf)
#pragma unroll
        for (int r = 0; r < 8; ++r) cc[hf][r] = 0.0f;

    __syncthreads();

    // ---------------- char LSTM: 20 sequential steps ----------------
#pragma unroll 1
    for (int t = 0; t < C_; ++t) {
        // Embedding lookup -> x tile (4 elements / thread)
#pragma unroll
        for (int it = 0; it < 4; ++it) {
            int idx = tid * 4 + it;
            int row = idx >> 6, e = idx & 63;
            int cid = char_ids[(size_t)(rowBase + row) * C_ + t];
            xbuf[idx] = embed[cid * 64 + e];
        }
        __syncthreads();

        v8f zero = {};
        v8f acc[4][2];
#pragma unroll
        for (int g = 0; g < 4; ++g)
#pragma unroll
            for (int hf = 0; hf < 2; ++hf) acc[g][hf] = zero;

        // K-outer: kb 0..1 = input projection (x @ Wih_c.T, E padded to 64),
        //          kb 2..9 = recurrent (h @ Whh_c.T, K = 256).
#pragma unroll 1
        for (int kb = 0; kb < 10; ++kb) {
            Frag a;
            const u32x4* wb;
            int kk;
            if (kb < 2) { a = load_afrag(xbuf, 64,  kb,     lane); wb = wihc4; kk = kb; }
            else        { a = load_afrag(hbuf, 256, kb - 2, lane); wb = whhc4; kk = kb - 2; }
#pragma unroll
            for (int g = 0; g < 4; ++g)
#pragma unroll
                for (int hf = 0; hf < 2; ++hf) {
                    Frag bf = load_bfrag(wb, kk, 64, g * 16 + wave * 2 + hf, lane);
                    acc[g][hf] = __builtin_amdgcn_wmma_f32_16x16x32_bf16(
                            false, a.v, false, bf.v, (short)0, acc[g][hf], false, false);
                }
        }
        __syncthreads();   // all hbuf/xbuf reads done before pointwise writes

        // LSTM pointwise update, fully in registers; hbuf keeps frozen h.
        // Activations via hardware v_tanh_f32 (TRANS32).
#pragma unroll
        for (int hf = 0; hf < 2; ++hf) {
            int col = wave * 32 + hf * 16 + (lane & 15);
#pragma unroll
            for (int r = 0; r < 8; ++r) {
                int row  = (lane < 16) ? r : (r + 8);
                float gi = acc[0][hf][r] + bi[hf];
                float gf = acc[1][hf][r] + bff[hf];
                float gg = acc[2][hf][r] + bgg[hf];
                float go = acc[3][hf][r] + boo[hf];
                float cn = fast_sigm(gf) * cc[hf][r] + fast_sigm(gi) * fast_tanh(gg);
                float hn = fast_sigm(go) * fast_tanh(cn);
                if (t < lenbuf[row]) {
                    cc[hf][r] = cn;
                    hbuf[row * 256 + col] = f2bf(hn);   // else keep frozen value
                }
            }
        }
        __syncthreads();
    }

    // ---------------- token encoder: single LSTM step from zero state ----------------
    {
        v8f zero = {};
        v8f acc[4][2];
#pragma unroll
        for (int g = 0; g < 4; ++g)
#pragma unroll
            for (int hf = 0; hf < 2; ++hf) acc[g][hf] = zero;

#pragma unroll 1
        for (int kb = 0; kb < 8; ++kb) {
            Frag a = load_afrag(hbuf, 256, kb, lane);
#pragma unroll
            for (int g = 0; g < 4; ++g)
#pragma unroll
                for (int hf = 0; hf < 2; ++hf) {
                    Frag bf = load_bfrag(wiht4, kb, 64, g * 16 + wave * 2 + hf, lane);
                    acc[g][hf] = __builtin_amdgcn_wmma_f32_16x16x32_bf16(
                            false, a.v, false, bf.v, (short)0, acc[g][hf], false, false);
                }
        }
        __syncthreads();

#pragma unroll
        for (int hf = 0; hf < 2; ++hf) {
            int col = wave * 32 + hf * 16 + (lane & 15);
#pragma unroll
            for (int r = 0; r < 8; ++r) {
                int row  = (lane < 16) ? r : (r + 8);
                float gi = acc[0][hf][r] + b_t[col];
                float gg = acc[2][hf][r] + b_t[512 + col];
                float go = acc[3][hf][r] + b_t[768 + col];
                float c1 = fast_sigm(gi) * fast_tanh(gg);   // zero initial cell
                float h1 = fast_sigm(go) * fast_tanh(c1);
                hbuf[row * 256 + col] = f2bf(h1);
            }
        }
        __syncthreads();
    }

    // ---------------- classifier: h1 @ W_cls + b_cls -> emissions [T,B,K] ----------------
    if (wave < 2) {
        v8f a = {};
#pragma unroll 1
        for (int kb = 0; kb < 8; ++kb) {
            Frag af = load_afrag(hbuf, 256, kb, lane);
            Frag bf = load_bfrag(wcls4, kb, 2, wave, lane);
            a = __builtin_amdgcn_wmma_f32_16x16x32_bf16(
                    false, af.v, false, bf.v, (short)0, a, false, false);
        }
        int k = wave * 16 + (lane & 15);
        if (k < K_) {
            float bk = b_cls[k];
#pragma unroll
            for (int r = 0; r < 8; ++r) {
                int row  = (lane < 16) ? r : (r + 8);
                int grow = rowBase + row;
                int bb   = grow >> 8;      // grow = b*T + t, T = 256
                int tt   = grow & 255;
                emis[((size_t)tt * B_ + bb) * K_ + k] = a[r] + bk;
            }
        }
    }
}

// ---------------------------------------------------------------------------
// Viterbi decode: one wave per line. K=25 states (one per lane), score vector
// shared via LDS, u8 backpointers in scratch, lane 0 backtracks.
// ---------------------------------------------------------------------------
__global__ __launch_bounds__(32)
void viterbi(const float* __restrict__ emis, const int* __restrict__ tok_len,
             const float* __restrict__ start_trans, const float* __restrict__ end_trans,
             const float* __restrict__ trans, unsigned char* __restrict__ bps,
             float* __restrict__ out_tags, float* __restrict__ out_score)
{
    __shared__ float tr[K_ * K_];
    __shared__ float sc[32];
    __shared__ float fin[32];

    const int b    = blockIdx.x;
    const int lane = threadIdx.x;
    for (int i = lane; i < K_ * K_; i += 32) tr[i] = trans[i];
    const int len = tok_len[b];

    float score = -1e30f;
    if (lane < K_) score = start_trans[lane] + emis[(size_t)b * K_ + lane];  // t = 0
    sc[lane] = score;
    __syncthreads();

    for (int t = 1; t < T_; ++t) {
        if (lane < K_) {
            float best = sc[0] + tr[lane];
            int   arg  = 0;
            for (int kp = 1; kp < K_; ++kp) {
                float v = sc[kp] + tr[kp * K_ + lane];
                if (v > best) { best = v; arg = kp; }
            }
            float nxt = best + emis[((size_t)t * B_ + b) * K_ + lane];
            int bp = lane;                        // identity past sequence end
            if (t < len) { score = nxt; bp = arg; }
            bps[((size_t)(t - 1) * B_ + b) * K_ + lane] = (unsigned char)bp;
        }
        __syncthreads();
        sc[lane] = score;
        __syncthreads();
    }

    fin[lane] = (lane < K_) ? (score + end_trans[lane]) : -1e30f;
    __syncthreads();

    if (lane == 0) {
        float bv = fin[0]; int ba = 0;
        for (int k = 1; k < K_; ++k) if (fin[k] > bv) { bv = fin[k]; ba = k; }
        out_score[b] = bv;
        int tag = ba;
        out_tags[(size_t)(T_ - 1) * B_ + b] = ((T_ - 1) < len) ? (float)tag : 0.0f;
        for (int i = T_ - 2; i >= 0; --i) {
            tag = bps[((size_t)i * B_ + b) * K_ + tag];
            out_tags[(size_t)i * B_ + b] = (i < len) ? (float)tag : 0.0f;
        }
    }
}

// ---------------------------------------------------------------------------
extern "C" void kernel_launch(void* const* d_in, const int* in_sizes, int n_in,
                              void* d_out, int out_size, void* d_ws, size_t ws_size,
                              hipStream_t stream) {
    (void)in_sizes; (void)n_in; (void)out_size; (void)ws_size;

    const int*   char_ids      = (const int*)d_in[0];
    const int*   char_lengths  = (const int*)d_in[1];
    const int*   token_lengths = (const int*)d_in[2];
    const float* char_embed    = (const float*)d_in[3];
    const float* Wih_c         = (const float*)d_in[4];
    const float* Whh_c         = (const float*)d_in[5];
    const float* b_c           = (const float*)d_in[6];
    const float* Wih_t         = (const float*)d_in[7];
    /* d_in[8] = Whh_t (unused by reference encoder) */
    const float* b_t           = (const float*)d_in[9];
    const float* W_cls         = (const float*)d_in[10];
    const float* b_cls         = (const float*)d_in[11];
    const float* start_trans   = (const float*)d_in[12];
    const float* end_trans     = (const float*)d_in[13];
    const float* trans         = (const float*)d_in[14];

    // Workspace layout (bytes, all 16B aligned):
    char* ws = (char*)d_ws;
    unsigned short* wihc   = (unsigned short*)(ws + 0);        // 2*64 frags  = 131072 B
    unsigned short* whhc   = (unsigned short*)(ws + 131072);   // 8*64 frags  = 524288 B
    unsigned short* wiht   = (unsigned short*)(ws + 655360);   // 8*64 frags  = 524288 B
    unsigned short* wcls   = (unsigned short*)(ws + 1179648);  // 8*2  frags  = 16384 B
    unsigned short* embedb = (unsigned short*)(ws + 1196032);  // 128*64 bf16 = 16384 B
    float*          emis   = (float*)(ws + 1212416);           // T*B*K f32   = 6553600 B
    unsigned char*  bps    = (unsigned char*)(ws + 7766016);   // (T-1)*B*K   = 1632000 B

    // Weight pre-swizzle into WMMA B-fragments (bf16, lane-major).
    prep_frags<<<256, 256, 0, stream>>>(Wih_c, wihc, 2, 64, E_,  G4H, E_,  1);
    prep_frags<<<512, 256, 0, stream>>>(Whh_c, whhc, 8, 64, H_,  G4H, H_,  1);
    prep_frags<<<512, 256, 0, stream>>>(Wih_t, wiht, 8, 64, H_,  G4H, H_,  1);
    prep_frags<<<16,  256, 0, stream>>>(W_cls, wcls, 8, 2,  H_,  K_,  K_,  0);
    prep_embed<<<32,  256, 0, stream>>>(char_embed, embedb);

    // Encoder: 65536 rows / 16 per WG = 4096 workgroups, 8 waves each.
    ner_encoder<<<4096, 256, 0, stream>>>(char_ids, char_lengths, b_c, b_t, b_cls,
                                          wihc, whhc, wiht, wcls, embedb, emis);

    // CRF Viterbi decode + outputs: tags [T,B] (as float) then best scores [B].
    float* out_tags  = (float*)d_out;
    float* out_score = out_tags + (size_t)T_ * B_;
    viterbi<<<B_, 32, 0, stream>>>(emis, token_lengths, start_trans, end_trans, trans,
                                   bps, out_tags, out_score);
}
